// pointnet2_encoder_45406394253985
// MI455X (gfx1250) — matrix operand order repaired and checked by reading, not compiled, not run on hardware
//
#include <hip/hip_runtime.h>
#include <hip/hip_bf16.h>
#include <stdint.h>

// ---------------------------------------------------------------------------
// Types for CDNA5 WMMA (wave32): v_wmma_f32_16x16x32_bf16
// ---------------------------------------------------------------------------
typedef __attribute__((ext_vector_type(16))) __bf16 v16bf;
typedef __attribute__((ext_vector_type(8)))  float  v8f;

union AV { v16bf v; uint4 u[2]; };

__device__ __forceinline__ uint16_t f2bf(float f) {
  union { float f; uint32_t u; } v; v.f = f;
  uint32_t r = v.u + 0x7fffu + ((v.u >> 16) & 1u);  // round-to-nearest-even
  return (uint16_t)(r >> 16);
}
__device__ __forceinline__ float bf2f(uint16_t h) {
  union { uint32_t u; float f; } v; v.u = ((uint32_t)h) << 16;
  return v.f;
}

// ---------------------------------------------------------------------------
// Wave-level 16x16 output tile, K-loop in steps of 32 (bf16).
// A: [rows, lda] row-major (K contiguous)  -> per ISA A-layout:
//    lanes 0-15 hold M=r, K=kk..kk+7 (v0-3) and kk+16..kk+23 (v4-7);
//    lanes 16-31 the +8 K offset.  Both halves are contiguous b128 loads.
// WT: [np, ldw] row-major = W transposed -> same K order as A:
//    K=kk+hi*8..+7 and kk+hi*8+16..+7: two b128 loads.
// ---------------------------------------------------------------------------
__device__ __forceinline__ v8f wmma_tile(const uint16_t* A, int lda,
                                         const uint16_t* WT, int ldw,
                                         int m0, int n0, int kp, v8f acc) {
  const int lane = threadIdx.x & 31;
  const int hi   = lane >> 4;          // 0 or 1
  const int r    = lane & 15;
  const uint16_t* arow = A  + (size_t)(m0 + r) * lda + hi * 8;
  const uint16_t* brow = WT + (size_t)(n0 + r) * ldw + hi * 8;
  for (int kk = 0; kk < kp; kk += 32) {
    AV a, b;
    a.u[0] = *(const uint4*)(arow + kk);        // K = kk+hi*8   .. +7
    a.u[1] = *(const uint4*)(arow + kk + 16);   // K = kk+hi*8+16.. +7
    b.u[0] = *(const uint4*)(brow + kk);        // K = kk+hi*8   .. +7
    b.u[1] = *(const uint4*)(brow + kk + 16);   // K = kk+hi*8+16.. +7
    acc = __builtin_amdgcn_wmma_f32_16x16x32_bf16(false, a.v, false, b.v,
                                                  (short)0, acc, false, false);
  }
  return acc;
}

// C/D layout: vgpr i = row m0+i (lanes 0-15) / m0+8+i (lanes 16-31), col = lane%16
__device__ __forceinline__ void store_tile_relu(uint16_t* O, int ldo, int m0, int n0,
                                                v8f c, const uint16_t* bias) {
  const int lane = threadIdx.x & 31;
  const int r    = lane & 15;
  const int mb   = m0 + ((lane < 16) ? 0 : 8);
  const float bv = bf2f(bias[n0 + r]);
#pragma unroll
  for (int i = 0; i < 8; ++i) {
    float v = c[i] + bv;
    v = v > 0.0f ? v : 0.0f;                    // ReLU (all layers in reference)
    O[(size_t)(mb + i) * ldo + (n0 + r)] = f2bf(v);
  }
}

// One dense layer: O[rows,np] = relu(A[rows,kp] @ WT^T + bias), tiles over waves
__device__ __forceinline__ void gemm_layer(const uint16_t* A, int kp,
                                           const uint16_t* WT, const uint16_t* bias,
                                           uint16_t* O, int np, int rows) {
  const int wave   = threadIdx.x >> 5;
  const int nwaves = blockDim.x >> 5;
  const int nt = np >> 4;
  const int total = (rows >> 4) * nt;
  for (int t = wave; t < total; t += nwaves) {
    const int m0 = (t / nt) << 4;
    const int n0 = (t % nt) << 4;
    v8f acc = {};
    acc = wmma_tile(A, kp, WT, kp, m0, n0, kp, acc);
    store_tile_relu(O, np, m0, n0, acc, bias);
  }
}

// ---------------------------------------------------------------------------
// Small utility kernels
// ---------------------------------------------------------------------------
__global__ void cvt_bf_kernel(const float* __restrict__ src, uint16_t* __restrict__ dst,
                              int n) {
  int i = blockIdx.x * blockDim.x + threadIdx.x;
  if (i < n) dst[i] = f2bf(src[i]);
}

// WT[np,kp] = W[cin,cout]^T padded with zeros; bias padded with zeros
__global__ void prep_w_kernel(const float* __restrict__ W, const float* __restrict__ b,
                              int cin, int cout, int kp, int np,
                              uint16_t* __restrict__ WT, uint16_t* __restrict__ bp) {
  int idx = blockIdx.x * blockDim.x + threadIdx.x;
  if (idx >= np * kp) return;
  int n = idx / kp, k = idx % kp;
  WT[idx] = (n < cout && k < cin) ? f2bf(W[(size_t)k * cout + n]) : (uint16_t)0;
  if (k == 0) bp[n] = (n < cout) ? f2bf(b[n]) : (uint16_t)0;
}

// Farthest point sampling; one block per batch; mind[] lives in dynamic LDS
__global__ void fps_kernel(const float* __restrict__ pts, int* __restrict__ fidx,
                           int N, int npoint) {
  extern __shared__ char smem[];
  float* mind = (float*)smem;
  float* rv   = (float*)(smem + (size_t)N * 4);
  int*   ri   = (int*)(smem + (size_t)N * 4 + (size_t)blockDim.x * 4);
  const int b = blockIdx.x, tid = threadIdx.x, bd = blockDim.x;
  const float* px = pts + (size_t)b * N * 3;
  for (int i = tid; i < N; i += bd) mind[i] = 1e10f;
  __syncthreads();
  int far = 0;
  for (int it = 0; it < npoint; ++it) {
    if (tid == 0) fidx[b * npoint + it] = far;
    const float cx = px[far * 3], cy = px[far * 3 + 1], cz = px[far * 3 + 2];
    float bestv = -1.0f; int besti = 0;
    for (int i = tid; i < N; i += bd) {
      float dx = px[i * 3] - cx, dy = px[i * 3 + 1] - cy, dz = px[i * 3 + 2] - cz;
      float m = fminf(mind[i], dx * dx + dy * dy + dz * dz);
      mind[i] = m;
      if (m > bestv) { bestv = m; besti = i; }
    }
    rv[tid] = bestv; ri[tid] = besti;
    __syncthreads();
    for (int s = bd >> 1; s > 0; s >>= 1) {
      if (tid < s) {
        if (rv[tid + s] > rv[tid] ||
            (rv[tid + s] == rv[tid] && ri[tid + s] < ri[tid])) {
          rv[tid] = rv[tid + s]; ri[tid] = ri[tid + s];
        }
      }
      __syncthreads();
    }
    far = ri[0];
    __syncthreads();
  }
}

__global__ void gather_xyz_kernel(const float* __restrict__ pts, const int* __restrict__ fidx,
                                  float* __restrict__ out, int N, int S, int total) {
  int idx = blockIdx.x * blockDim.x + threadIdx.x;
  if (idx >= total) return;
  int d = idx % 3, rs = idx / 3, b = rs / S;
  out[idx] = pts[((size_t)b * N + fidx[rs]) * 3 + d];
}

__device__ __forceinline__ void fill_row(uint16_t* row, int k0p,
                                         const uint16_t* pf, int j, int C,
                                         float dx, float dy, float dz) {
  row[0] = f2bf(dx); row[1] = f2bf(dy); row[2] = f2bf(dz);
  const uint16_t* f = pf + (size_t)j * C;
  for (int c = 0; c < C; ++c) row[3 + c] = f[c];
  for (int c = 3 + C; c < k0p; ++c) row[c] = 0;   // zero pad cols
}

// ---------------------------------------------------------------------------
// Fused set-abstraction kernel: one block per (b,s) group.
// Stage 1: wave0 ordered ball-query -> bf16 rows [dx,dy,dz,feats,pad] in LDS.
// Stage 2: 3 WMMA GEMM+ReLU layers ping-ponging LDS (A from LDS, WT from L2).
// Stage 3: channel-wise max over K rows -> outf[b*S+s, coff..coff+cout)
// ---------------------------------------------------------------------------
__global__ void __launch_bounds__(256)
sa_fused_kernel(const float* __restrict__ xyz, const float* __restrict__ nxyz,
                const uint16_t* __restrict__ feats, int C, int N, int S, int K, float r2,
                const uint16_t* __restrict__ w0, const uint16_t* __restrict__ b0,
                int k0p, int n0p,
                const uint16_t* __restrict__ w1, const uint16_t* __restrict__ b1, int n1p,
                const uint16_t* __restrict__ w2, const uint16_t* __restrict__ b2, int n2p,
                int cmaxA, uint16_t* __restrict__ outf, int Cdst, int coff, int coutReal) {
  extern __shared__ char smem[];
  uint16_t* bufA = (uint16_t*)smem;
  uint16_t* bufB = bufA + (size_t)K * cmaxA;
  const int gs = blockIdx.x;           // b*S + s
  const int b  = gs / S;
  const int tid = threadIdx.x, lane = tid & 31, wave = tid >> 5;

  if (wave == 0) {  // ball query: first K hits in index order, pad with first hit
    const float cx = nxyz[gs * 3 + 0], cy = nxyz[gs * 3 + 1], cz = nxyz[gs * 3 + 2];
    const float* px = xyz + (size_t)b * N * 3;
    const uint16_t* pf = feats + (size_t)b * N * C;
    int count = 0, first = -1;
    for (int base = 0; base < N && count < K; base += 32) {
      const int j = base + lane;
      float dx = 0.f, dy = 0.f, dz = 0.f;
      bool in = false;
      if (j < N) {
        dx = px[j * 3 + 0] - cx; dy = px[j * 3 + 1] - cy; dz = px[j * 3 + 2] - cz;
        in = (dx * dx + dy * dy + dz * dz) <= r2;
      }
      const unsigned m = (unsigned)__ballot(in);
      if (first < 0 && m) first = base + __builtin_ctz(m);
      const int slot = count + __builtin_popcount(m & ((1u << lane) - 1u));
      if (in && slot < K) fill_row(bufA + (size_t)slot * k0p, k0p, pf, j, C, dx, dy, dz);
      count += __builtin_popcount(m);
    }
    if (first < 0) first = 0;
    if (count < K) {
      const float dx = px[first * 3 + 0] - cx;
      const float dy = px[first * 3 + 1] - cy;
      const float dz = px[first * 3 + 2] - cz;
      for (int slot = count + lane; slot < K; slot += 32)
        fill_row(bufA + (size_t)slot * k0p, k0p, pf, first, C, dx, dy, dz);
    }
  }
  __syncthreads();
  gemm_layer(bufA, k0p, w0, b0, bufB, n0p, K);
  __syncthreads();
  gemm_layer(bufB, n0p, w1, b1, bufA, n1p, K);
  __syncthreads();
  gemm_layer(bufA, n1p, w2, b2, bufB, n2p, K);
  __syncthreads();
  uint16_t* orow = outf + (size_t)gs * Cdst + coff;
  for (int c = tid; c < coutReal; c += blockDim.x) {
    float m = -3.4e38f;
    for (int k = 0; k < K; ++k) m = fmaxf(m, bf2f(bufB[(size_t)k * n2p + c]));
    orow[c] = f2bf(m);
  }
}

// sa3 input: rows [B*128, 544] = [nxyz2 (3) | l2 (512) | zero pad (29)]
__global__ void sa3in_kernel(const float* __restrict__ nxyz2, const uint16_t* __restrict__ l2,
                             uint16_t* __restrict__ out) {
  int idx = blockIdx.x * blockDim.x + threadIdx.x;
  if (idx >= 1024 * 544) return;
  int row = idx / 544, c = idx % 544;
  uint16_t v = 0;
  if (c < 3)        v = f2bf(nxyz2[row * 3 + c]);
  else if (c < 515) v = l2[(size_t)row * 512 + (c - 3)];
  out[idx] = v;
}

// Unfused global GEMM (one 16x16 tile per wave) for sa3
__global__ void gemm_kernel(const uint16_t* __restrict__ A, int M, int kp,
                            const uint16_t* __restrict__ WT, const uint16_t* __restrict__ bias,
                            uint16_t* __restrict__ O, int np) {
  const int wave = (int)((blockIdx.x * blockDim.x + threadIdx.x) >> 5);
  const int nt = np >> 4;
  const int total = (M >> 4) * nt;
  if (wave >= total) return;
  const int m0 = (wave / nt) << 4, n0 = (wave % nt) << 4;
  v8f acc = {};
  acc = wmma_tile(A, kp, WT, kp, m0, n0, kp, acc);
  store_tile_relu(O, np, m0, n0, acc, bias);
}

__global__ void final_max_kernel(const uint16_t* __restrict__ act, float* __restrict__ out) {
  int idx = blockIdx.x * blockDim.x + threadIdx.x;
  if (idx >= 8 * 1024) return;
  int b = idx >> 10, c = idx & 1023;
  float m = -3.4e38f;
  for (int s = 0; s < 128; ++s)
    m = fmaxf(m, bf2f(act[((size_t)(b * 128 + s) << 10) + c]));
  out[idx] = m;
}

// ---------------------------------------------------------------------------
// Host orchestration
// ---------------------------------------------------------------------------
extern "C" void kernel_launch(void* const* d_in, const int* in_sizes, int n_in,
                              void* d_out, int out_size, void* d_ws, size_t ws_size,
                              hipStream_t stream) {
  (void)in_sizes; (void)n_in; (void)out_size; (void)ws_size;
  const int Bn = 8, Npts = 16384;
  const float* xyz = (const float*)d_in[0];

  char* ws = (char*)d_ws;
  size_t off = 0;
  auto alloc = [&](size_t bytes) -> char* {
    char* p = ws + off;
    off = (off + bytes + 255) & ~(size_t)255;
    return p;
  };
  uint16_t* xyzbf = (uint16_t*)alloc((size_t)Bn * Npts * 3 * 2);
  int*      fidx1 = (int*)     alloc((size_t)Bn * 512 * 4);
  float*    nxyz1 = (float*)   alloc((size_t)Bn * 512 * 3 * 4);
  uint16_t* l1    = (uint16_t*)alloc((size_t)Bn * 512 * 320 * 2);
  int*      fidx2 = (int*)     alloc((size_t)Bn * 128 * 4);
  float*    nxyz2 = (float*)   alloc((size_t)Bn * 128 * 3 * 4);
  uint16_t* l2    = (uint16_t*)alloc((size_t)Bn * 128 * 512 * 2);
  uint16_t* sa3in = (uint16_t*)alloc((size_t)1024 * 544 * 2);
  uint16_t* g0    = (uint16_t*)alloc((size_t)1024 * 256 * 2);
  uint16_t* g1    = (uint16_t*)alloc((size_t)1024 * 512 * 2);
  uint16_t* g2    = (uint16_t*)alloc((size_t)1024 * 1024 * 2);

  // layer table: cin, cout, d_in index of W (bias at +1); flatten order:
  // sa1[3 mlps x 3 layers], sa2[2 x 3], sa3[3]
  struct LD { int cin, cout, win; };
  const LD L[18] = {
    {6, 32, 1}, {32, 32, 3}, {32, 64, 5},
    {6, 64, 7}, {64, 64, 9}, {64, 128, 11},
    {6, 64, 13}, {64, 96, 15}, {96, 128, 17},
    {323, 128, 19}, {128, 128, 21}, {128, 256, 23},
    {323, 128, 25}, {128, 196, 27}, {196, 256, 29},
    {515, 256, 31}, {256, 512, 33}, {512, 1024, 35},
  };
  uint16_t* WT[18]; uint16_t* BP[18]; int KP[18], NP[18];
  for (int i = 0; i < 18; ++i) {
    KP[i] = (L[i].cin + 31) & ~31;
    NP[i] = (L[i].cout + 31) & ~31;
    WT[i] = (uint16_t*)alloc((size_t)NP[i] * KP[i] * 2);
    BP[i] = (uint16_t*)alloc((size_t)NP[i] * 2);
  }

  {  // xyz -> bf16 features for sa1
    int n = Bn * Npts * 3;
    cvt_bf_kernel<<<(n + 255) / 256, 256, 0, stream>>>(xyz, xyzbf, n);
  }
  for (int i = 0; i < 18; ++i) {  // pre-transpose + pad all weights (bf16)
    int tot = NP[i] * KP[i];
    prep_w_kernel<<<(tot + 255) / 256, 256, 0, stream>>>(
        (const float*)d_in[L[i].win], (const float*)d_in[L[i].win + 1],
        L[i].cin, L[i].cout, KP[i], NP[i], WT[i], BP[i]);
  }

  // ----- SA1 -----
  fps_kernel<<<Bn, 256, (size_t)Npts * 4 + 256 * 8, stream>>>(xyz, fidx1, Npts, 512);
  {
    int tot = Bn * 512 * 3;
    gather_xyz_kernel<<<(tot + 255) / 256, 256, 0, stream>>>(xyz, fidx1, nxyz1, Npts, 512, tot);
  }
  {
    const float rr[3] = {0.1f, 0.2f, 0.4f};
    const int KK[3] = {32, 64, 128};
    const int coffs[3] = {0, 64, 192};
    for (int sc = 0; sc < 3; ++sc) {
      int Lb = sc * 3;
      int k0p = KP[Lb], n0p = NP[Lb], n1p = NP[Lb + 1], n2p = NP[Lb + 2];
      int cmaxA = k0p > n1p ? k0p : n1p;
      int cmaxB = n0p > n2p ? n0p : n2p;
      size_t shmem = (size_t)KK[sc] * (cmaxA + cmaxB) * 2;
      sa_fused_kernel<<<Bn * 512, 256, shmem, stream>>>(
          xyz, nxyz1, xyzbf, 3, Npts, 512, KK[sc], rr[sc] * rr[sc],
          WT[Lb], BP[Lb], k0p, n0p,
          WT[Lb + 1], BP[Lb + 1], n1p,
          WT[Lb + 2], BP[Lb + 2], n2p,
          cmaxA, l1, 320, coffs[sc], L[Lb + 2].cout);
    }
  }

  // ----- SA2 -----
  fps_kernel<<<Bn, 256, (size_t)512 * 4 + 256 * 8, stream>>>(nxyz1, fidx2, 512, 128);
  {
    int tot = Bn * 128 * 3;
    gather_xyz_kernel<<<(tot + 255) / 256, 256, 0, stream>>>(nxyz1, fidx2, nxyz2, 512, 128, tot);
  }
  {
    const float rr[2] = {0.4f, 0.8f};
    const int KK[2] = {64, 128};
    const int coffs[2] = {0, 256};
    for (int sc = 0; sc < 2; ++sc) {
      int Lb = 9 + sc * 3;
      int k0p = KP[Lb], n0p = NP[Lb], n1p = NP[Lb + 1], n2p = NP[Lb + 2];
      int cmaxA = k0p > n1p ? k0p : n1p;
      int cmaxB = n0p > n2p ? n0p : n2p;
      size_t shmem = (size_t)KK[sc] * (cmaxA + cmaxB) * 2;
      sa_fused_kernel<<<Bn * 128, 256, shmem, stream>>>(
          nxyz1, nxyz2, l1, 320, 512, 128, KK[sc], rr[sc] * rr[sc],
          WT[Lb], BP[Lb], k0p, n0p,
          WT[Lb + 1], BP[Lb + 1], n1p,
          WT[Lb + 2], BP[Lb + 2], n2p,
          cmaxA, l2, 512, coffs[sc], L[Lb + 2].cout);
    }
  }

  // ----- SA3 (group_all) -----
  {
    int tot = 1024 * 544;
    sa3in_kernel<<<(tot + 255) / 256, 256, 0, stream>>>(nxyz2, l2, sa3in);
  }
  {
    int waves = (1024 / 16) * (NP[15] / 16);
    gemm_kernel<<<(waves * 32 + 255) / 256, 256, 0, stream>>>(sa3in, 1024, KP[15], WT[15], BP[15], g0, NP[15]);
  }
  {
    int waves = (1024 / 16) * (NP[16] / 16);
    gemm_kernel<<<(waves * 32 + 255) / 256, 256, 0, stream>>>(g0, 1024, KP[16], WT[16], BP[16], g1, NP[16]);
  }
  {
    int waves = (1024 / 16) * (NP[17] / 16);
    gemm_kernel<<<(waves * 32 + 255) / 256, 256, 0, stream>>>(g1, 1024, KP[17], WT[17], BP[17], g2, NP[17]);
  }
  final_max_kernel<<<(8 * 1024 + 255) / 256, 256, 0, stream>>>(g2, (float*)d_out);
}
